// STG_GNN_22677427323641
// MI455X (gfx1250) — matrix operand order repaired and checked by reading, not compiled, not run on hardware
//
#include <hip/hip_runtime.h>
#include <hip/hip_bf16.h>
#include <math.h>

#define B_SZ 8192
#define FIN  160
#define NNB  35
#define NG   8
#define TT   52
#define HH   16

typedef __attribute__((ext_vector_type(16))) __bf16 v16bf;
typedef __attribute__((ext_vector_type(8)))  float  v8f;
typedef __attribute__((ext_vector_type(4)))  float  f4;

union V16 {
  v16bf bf;
  __bf16 h[16];
  unsigned short us[16];
  uint4 q[2];
};

__device__ __forceinline__ unsigned short bf16bits(float f) {
  union { __bf16 h; unsigned short u; } c;
  c.h = (__bf16)f;                      // native f32 -> bf16 (RNE)
  return c.u;
}

__device__ __forceinline__ void pack8(const float* v, __bf16* b) {
  #pragma unroll
  for (int j = 0; j < 8; ++j) b[j] = (__bf16)v[j];
}

// ---------------------------------------------------------------------------
// Kernel 1: GRU (week + work) + temporal FC.  One thread per batch row.
// ---------------------------------------------------------------------------
__global__ void gru_kernel(const float* __restrict__ his,
    const float* wk_wih, const float* wk_whh, const float* wk_bih, const float* wk_bhh,
    const float* wk_ow,  const float* wk_ob,
    const float* wo_wih, const float* wo_whh, const float* wo_bih, const float* wo_bhh,
    const float* wo_ow,  const float* wo_ob,
    const float* tfw, const float* tfb, float* __restrict__ temp_pred)
{
  __shared__ float s_whh[2][48*HH];
  __shared__ float s_wih[2][48], s_bih[2][48], s_bhh[2][48];
  __shared__ float s_ow[2][48], s_ob[2][3];
  __shared__ float s_tfw[12], s_tfb[2];
  int tid = threadIdx.x;
  for (int i = tid; i < 48*HH; i += blockDim.x) { s_whh[0][i] = wk_whh[i]; s_whh[1][i] = wo_whh[i]; }
  for (int i = tid; i < 48; i += blockDim.x) {
    s_wih[0][i] = wk_wih[i]; s_wih[1][i] = wo_wih[i];
    s_bih[0][i] = wk_bih[i]; s_bih[1][i] = wo_bih[i];
    s_bhh[0][i] = wk_bhh[i]; s_bhh[1][i] = wo_bhh[i];
    s_ow[0][i]  = wk_ow[i];  s_ow[1][i]  = wo_ow[i];
  }
  if (tid < 3)  { s_ob[0][tid] = wk_ob[tid]; s_ob[1][tid] = wo_ob[tid]; }
  if (tid < 12) s_tfw[tid] = tfw[tid];
  if (tid < 2)  s_tfb[tid] = tfb[tid];
  __syncthreads();

  int b = blockIdx.x * blockDim.x + tid;
  if (b >= B_SZ) return;

  float xt6[6];
  for (int br = 0; br < 2; ++br) {
    float h[HH];
    #pragma unroll
    for (int i = 0; i < HH; ++i) h[i] = 0.f;
    const float* xs = his + (size_t)b * (2*TT) + br*TT;
    for (int t = 0; t < TT; ++t) {
      float x = xs[t];
      float nh[HH];
      #pragma unroll
      for (int i = 0; i < HH; ++i) {
        float ghr = s_bhh[br][i], ghz = s_bhh[br][HH+i], ghn = s_bhh[br][2*HH+i];
        #pragma unroll
        for (int k = 0; k < HH; ++k) {
          ghr += h[k] * s_whh[br][i*HH + k];
          ghz += h[k] * s_whh[br][(HH+i)*HH + k];
          ghn += h[k] * s_whh[br][(2*HH+i)*HH + k];
        }
        float gir = x*s_wih[br][i]      + s_bih[br][i];
        float giz = x*s_wih[br][HH+i]   + s_bih[br][HH+i];
        float gin = x*s_wih[br][2*HH+i] + s_bih[br][2*HH+i];
        float r = 1.f/(1.f + __expf(-(gir+ghr)));
        float z = 1.f/(1.f + __expf(-(giz+ghz)));
        float e2 = __expf(2.f*(gin + r*ghn));
        float nn = (e2-1.f)/(e2+1.f);
        nh[i] = (1.f - z)*nn + z*h[i];
      }
      #pragma unroll
      for (int i = 0; i < HH; ++i) h[i] = nh[i];
    }
    #pragma unroll
    for (int j = 0; j < 3; ++j) {
      float s = s_ob[br][j];
      #pragma unroll
      for (int k = 0; k < HH; ++k) s += h[k] * s_ow[br][j*HH + k];
      xt6[br*3 + j] = s;
    }
  }
  #pragma unroll
  for (int c = 0; c < 2; ++c) {
    float s = s_tfb[c];
    #pragma unroll
    for (int j = 0; j < 6; ++j) s += xt6[j] * s_tfw[c*6 + j];
    temp_pred[b*2 + c] = s;
  }
}

// ---------------------------------------------------------------------------
// Kernel 2: neighbor align GEMM via WMMA bf16.
// X: [M,160] f32 (M = B*35) streamed once -> non-temporal loads.
// B fragments pre-packed (bf16, WMMA lane order) in LDS once per block.
// ---------------------------------------------------------------------------
__global__ void align_neigh_kernel(const float* __restrict__ X,
                                   const float* __restrict__ W,
                                   const float* __restrict__ bias,
                                   float* __restrict__ out)
{
  __shared__ __align__(16) unsigned short sB[5][32][16];   // 5 KB
  int tid = threadIdx.x;
  for (int idx = tid; idx < 5*32*16; idx += blockDim.x) {
    int j  = idx & 15;
    int ln = (idx >> 4) & 31;
    int ks = idx >> 9;
    int hf = ln >> 4, cl = ln & 15;
    int k  = ks*32 + (j >> 3)*16 + hf*8 + (j & 7);
    float w = W[(cl & 7)*FIN + k];
    if (cl >= NG) w = 0.f;
    sB[ks][ln][j] = bf16bits(w);
  }
  __syncthreads();

  int wid  = blockIdx.x * (blockDim.x >> 5) + (tid >> 5);
  int lane = tid & 31;
  int half = lane >> 4;
  int mrow = lane & 15;
  int col  = lane & 15;
  size_t m0 = (size_t)wid * 16;

  const float* arow = X + (m0 + mrow) * FIN;
  v8f acc = {};
  #pragma unroll
  for (int ks = 0; ks < 5; ++ks) {
    int kbase = ks * 32;
    const f4* p = (const f4*)(arow + kbase + half*8);
    f4 c0 = __builtin_nontemporal_load(p);
    f4 c1 = __builtin_nontemporal_load(p + 1);
    f4 c2 = __builtin_nontemporal_load(p + 4);
    f4 c3 = __builtin_nontemporal_load(p + 5);
    float lo[8] = {c0.x,c0.y,c0.z,c0.w,c1.x,c1.y,c1.z,c1.w};
    float hi[8] = {c2.x,c2.y,c2.z,c2.w,c3.x,c3.y,c3.z,c3.w};
    V16 a;
    pack8(lo, a.h);
    pack8(hi, a.h + 8);
    V16 bw;
    bw.q[0] = *(const uint4*)&sB[ks][lane][0];
    bw.q[1] = *(const uint4*)&sB[ks][lane][8];
    acc = __builtin_amdgcn_wmma_f32_16x16x32_bf16(false, a.bf, false, bw.bf,
                                                  (short)0, acc, false, false);
  }
  if (col < NG) {
    float bb = bias[col];
    #pragma unroll
    for (int r = 0; r < 8; ++r)
      out[(m0 + r + half*8) * NG + col] = acc[r] + bb;
  }
}

// ---------------------------------------------------------------------------
// Kernel 3: target align + GAT attention (online softmax) + aggregate + ELU.
// ---------------------------------------------------------------------------
__global__ void graph_attn_kernel(const float* __restrict__ x_od,
    const float* aw, const float* ab, const float* a1w, const float* a1b,
    const float* a2w, const float* a2b,
    const float* __restrict__ nAll, float* __restrict__ gout)
{
  __shared__ float sAW[NG*FIN];
  __shared__ float sAB[NG], sA1[256], sB1[16], sA2[16], sB2;
  int tid = threadIdx.x;
  for (int i = tid; i < NG*FIN; i += blockDim.x) sAW[i] = aw[i];
  for (int i = tid; i < 256;    i += blockDim.x) sA1[i] = a1w[i];
  if (tid < NG) sAB[tid] = ab[tid];
  if (tid < 16) { sB1[tid] = a1b[tid]; sA2[tid] = a2w[tid]; }
  if (tid == 0) sB2 = a2b[0];
  __syncthreads();

  int b = blockIdx.x * blockDim.x + tid;
  if (b >= B_SZ) return;

  const float* xr = x_od + (size_t)b * FIN;
  float t8[NG];
  #pragma unroll
  for (int o = 0; o < NG; ++o) t8[o] = sAB[o];
  for (int k = 0; k < FIN; ++k) {
    float xv = xr[k];
    #pragma unroll
    for (int o = 0; o < NG; ++o) t8[o] += xv * sAW[o*FIN + k];
  }

  const float* nrow = nAll + (size_t)b * NNB * NG;
  float m = -3.0e38f, s = 0.f, acc[NG];
  #pragma unroll
  for (int o = 0; o < NG; ++o) acc[o] = 0.f;

  for (int j = 0; j < NNB; ++j) {
    float nj[NG];
    #pragma unroll
    for (int o = 0; o < NG; ++o) nj[o] = nrow[j*NG + o];
    float e = sB2;
    #pragma unroll
    for (int u = 0; u < 16; ++u) {
      float v = sB1[u];
      #pragma unroll
      for (int q = 0; q < NG; ++q) v += t8[q] * sA1[u*16 + q];
      #pragma unroll
      for (int q = 0; q < NG; ++q) v += nj[q] * sA1[u*16 + NG + q];
      v = v > 0.f ? v : 0.f;
      e += sA2[u] * v;
    }
    e = e > 0.f ? e : 0.2f*e;                     // leaky_relu(0.2)
    float nm = m > e ? m : e;
    float cs = __expf(m - nm);
    float ce = __expf(e - nm);
    s = s*cs + ce;
    #pragma unroll
    for (int o = 0; o < NG; ++o) acc[o] = acc[o]*cs + ce*nj[o];
    m = nm;
  }
  float inv = 1.f / s;
  #pragma unroll
  for (int o = 0; o < NG; ++o) {
    float hv = acc[o] * inv;
    gout[(size_t)b*NG + o] = hv > 0.f ? hv : (__expf(hv) - 1.f);   // ELU
  }
}

// ---------------------------------------------------------------------------
// Kernel 4: fc1 = relu([x | g_time | g_link | g_geo] @ W^T + b), WMMA bf16.
// grid: 512 blocks; blockIdx.x = mtGroup (0..127) + 128*grp (0..3).
// 4 waves/block share one N-group (4 N-tiles) -> B staged once in LDS (24KB).
// ---------------------------------------------------------------------------
__device__ __forceinline__ void fc1_load8(const float* __restrict__ X,
    const float* __restrict__ gt, const float* __restrict__ gl,
    const float* __restrict__ gg, size_t row, int k0, float* v)
{
  const f4* p;
  if (k0 < 160)       p = (const f4*)(X  + row*160 + k0);
  else if (k0 == 160) p = (const f4*)(gt + row*8);
  else if (k0 == 168) p = (const f4*)(gl + row*8);
  else if (k0 == 176) p = (const f4*)(gg + row*8);
  else {
    #pragma unroll
    for (int j = 0; j < 8; ++j) v[j] = 0.f;
    return;
  }
  f4 a = p[0], b = p[1];
  v[0]=a.x; v[1]=a.y; v[2]=a.z; v[3]=a.w; v[4]=b.x; v[5]=b.y; v[6]=b.z; v[7]=b.w;
}

__global__ void fc1_kernel(const float* __restrict__ X,
    const float* __restrict__ gt, const float* __restrict__ gl, const float* __restrict__ gg,
    const float* __restrict__ W, const float* __restrict__ bias,
    float* __restrict__ out)
{
  __shared__ __align__(16) unsigned short sB[6][4][32][16];   // 24 KB
  int tid = threadIdx.x;
  int grp = blockIdx.x >> 7;            // 0..3
  int mtG = blockIdx.x & 127;           // 0..127

  for (int idx = tid; idx < 6*4*32*16; idx += blockDim.x) {
    int j  = idx & 15;
    int ln = (idx >> 4) & 31;
    int nt = (idx >> 9) & 3;
    int ks = idx >> 11;
    int hf = ln >> 4, cl = ln & 15;
    int colg = (grp*4 + nt)*16 + cl;
    int k = ks*32 + (j >> 3)*16 + hf*8 + (j & 7);
    int kc = k < 184 ? k : 183;
    float w = W[(size_t)colg*184 + kc];
    if (k >= 184) w = 0.f;
    sB[ks][nt][ln][j] = bf16bits(w);
  }
  __syncthreads();

  int wv   = tid >> 5;
  int lane = tid & 31;
  int half = lane >> 4, mrow = lane & 15, col = lane & 15;
  int mtile = mtG*4 + wv;
  size_t row = (size_t)mtile*16 + mrow;

  v8f acc[4] = {};
  #pragma unroll
  for (int ks = 0; ks < 6; ++ks) {
    int kbase = ks * 32;
    float lo[8], hi[8];
    fc1_load8(X, gt, gl, gg, row, kbase + half*8,      lo);
    fc1_load8(X, gt, gl, gg, row, kbase + half*8 + 16, hi);
    V16 a;
    pack8(lo, a.h);
    pack8(hi, a.h + 8);
    #pragma unroll
    for (int nt = 0; nt < 4; ++nt) {
      V16 bw;
      bw.q[0] = *(const uint4*)&sB[ks][nt][lane][0];
      bw.q[1] = *(const uint4*)&sB[ks][nt][lane][8];
      acc[nt] = __builtin_amdgcn_wmma_f32_16x16x32_bf16(false, a.bf, false, bw.bf,
                                                        (short)0, acc[nt], false, false);
    }
  }
  #pragma unroll
  for (int nt = 0; nt < 4; ++nt) {
    int colg = (grp*4 + nt)*16 + col;
    float bb = bias[colg];
    #pragma unroll
    for (int r = 0; r < 8; ++r) {
      float v = acc[nt][r] + bb;
      out[((size_t)mtile*16 + r + half*8)*256 + colg] = v > 0.f ? v : 0.f;
    }
  }
}

// ---------------------------------------------------------------------------
// Kernel 5: generic relu(X @ W^T + b), WMMA bf16.  K % 32 == 0, in-bounds N.
// ---------------------------------------------------------------------------
__global__ void fcrelu_kernel(const float* __restrict__ X, const float* __restrict__ W,
    const float* __restrict__ bias, float* __restrict__ out, int K, int N, int nGroups)
{
  int lw    = blockIdx.x * (blockDim.x >> 5) + (threadIdx.x >> 5);
  int mtile = lw / nGroups, grp = lw % nGroups;
  int lane  = threadIdx.x & 31, half = lane >> 4, mrow = lane & 15, col = lane & 15;
  const float* xr = X + ((size_t)mtile*16 + mrow) * K;

  v8f acc[4] = {};
  int ksteps = K >> 5;
  for (int ks = 0; ks < ksteps; ++ks) {
    int kbase = ks * 32;
    const f4* p = (const f4*)(xr + kbase + half*8);
    f4 c0 = p[0], c1 = p[1], c2 = p[4], c3 = p[5];
    float lo[8] = {c0.x,c0.y,c0.z,c0.w,c1.x,c1.y,c1.z,c1.w};
    float hi[8] = {c2.x,c2.y,c2.z,c2.w,c3.x,c3.y,c3.z,c3.w};
    V16 a;
    pack8(lo, a.h);
    pack8(hi, a.h + 8);
    #pragma unroll
    for (int nt = 0; nt < 4; ++nt) {
      int colg = (grp*4 + nt)*16 + col;
      const f4* wp = (const f4*)(W + (size_t)colg*K + kbase + half*8);
      f4 w0 = wp[0], w1 = wp[1], w2 = wp[4], w3 = wp[5];
      float wl[8] = {w0.x,w0.y,w0.z,w0.w,w1.x,w1.y,w1.z,w1.w};
      float wh[8] = {w2.x,w2.y,w2.z,w2.w,w3.x,w3.y,w3.z,w3.w};
      V16 bw;
      pack8(wl, bw.h);
      pack8(wh, bw.h + 8);
      acc[nt] = __builtin_amdgcn_wmma_f32_16x16x32_bf16(false, a.bf, false, bw.bf,
                                                        (short)0, acc[nt], false, false);
    }
  }
  #pragma unroll
  for (int nt = 0; nt < 4; ++nt) {
    int colg = (grp*4 + nt)*16 + col;
    float bb = bias[colg];
    #pragma unroll
    for (int r = 0; r < 8; ++r) {
      float v = acc[nt][r] + bb;
      out[((size_t)mtile*16 + r + half*8)*N + colg] = v > 0.f ? v : 0.f;
    }
  }
}

// ---------------------------------------------------------------------------
// Kernel 6: fc3 + age gate + blend.  One thread per batch row.
// ---------------------------------------------------------------------------
__global__ void head_kernel(const float* __restrict__ h2,
    const float* f3w, const float* f3b,
    const float* __restrict__ od_age,
    const float* g1w, const float* g1b, const float* g2w, const float* g2b,
    const float* __restrict__ temp_pred, float* __restrict__ out)
{
  __shared__ float sW[2*128], sB[2], sG1w[16], sG1b[16], sG2w[32], sG2b[2];
  int tid = threadIdx.x;
  for (int i = tid; i < 256; i += blockDim.x) sW[i] = f3w[i];
  if (tid < 2)  { sB[tid] = f3b[tid]; sG2b[tid] = g2b[tid]; }
  if (tid < 16) { sG1w[tid] = g1w[tid]; sG1b[tid] = g1b[tid]; }
  if (tid < 32) sG2w[tid] = g2w[tid];
  __syncthreads();

  int b = blockIdx.x * blockDim.x + tid;
  if (b >= B_SZ) return;

  const float* hr = h2 + (size_t)b * 128;
  float p0 = sB[0], p1 = sB[1];
  for (int k = 0; k < 128; ++k) { float v = hr[k]; p0 += v*sW[k]; p1 += v*sW[128+k]; }

  float age = od_age[b];
  float l0 = sG2b[0], l1 = sG2b[1];
  #pragma unroll
  for (int u = 0; u < 16; ++u) {
    float g = age*sG1w[u] + sG1b[u];
    g = g > 0.f ? g : 0.f;
    l0 += g*sG2w[u];
    l1 += g*sG2w[16+u];
  }
  float mx = l0 > l1 ? l0 : l1;
  float e0 = __expf(l0 - mx), e1 = __expf(l1 - mx);
  float inv = 1.f / (e0 + e1);
  float gate0 = e0*inv, gate1 = e1*inv;
  out[b*2+0] = gate0*temp_pred[b*2+0] + gate1*p0;
  out[b*2+1] = gate0*temp_pred[b*2+1] + gate1*p1;
}

// ---------------------------------------------------------------------------
// Host launch.  Param pointers follow jax pytree flattening (dict keys sorted
// alphabetically at each nesting level) after the 6 data inputs.
// ---------------------------------------------------------------------------
extern "C" void kernel_launch(void* const* d_in, const int* in_sizes, int n_in,
                              void* d_out, int out_size, void* d_ws, size_t ws_size,
                              hipStream_t stream) {
  (void)in_sizes; (void)n_in; (void)out_size; (void)ws_size;
  const float* x_od    = (const float*)d_in[0];
  const float* his     = (const float*)d_in[1];
  const float* nb_geo  = (const float*)d_in[2];
  const float* nb_link = (const float*)d_in[3];
  const float* nb_time = (const float*)d_in[4];
  const float* age     = (const float*)d_in[5];
  const float* fc1_b = (const float*)d_in[6];
  const float* fc1_w = (const float*)d_in[7];
  const float* fc2_b = (const float*)d_in[8];
  const float* fc2_w = (const float*)d_in[9];
  const float* fc3_b = (const float*)d_in[10];
  const float* fc3_w = (const float*)d_in[11];
  const float* g1_b  = (const float*)d_in[12];
  const float* g1_w  = (const float*)d_in[13];
  const float* g2_b  = (const float*)d_in[14];
  const float* g2_w  = (const float*)d_in[15];
  const float* geo_ab = (const float*)d_in[16];
  const float* geo_aw = (const float*)d_in[17];
  const float* geo_1b = (const float*)d_in[18];
  const float* geo_1w = (const float*)d_in[19];
  const float* geo_2b = (const float*)d_in[20];
  const float* geo_2w = (const float*)d_in[21];
  const float* lnk_ab = (const float*)d_in[22];
  const float* lnk_aw = (const float*)d_in[23];
  const float* lnk_1b = (const float*)d_in[24];
  const float* lnk_1w = (const float*)d_in[25];
  const float* lnk_2b = (const float*)d_in[26];
  const float* lnk_2w = (const float*)d_in[27];
  const float* tf_b   = (const float*)d_in[28];
  const float* tf_w   = (const float*)d_in[29];
  const float* tim_ab = (const float*)d_in[30];
  const float* tim_aw = (const float*)d_in[31];
  const float* tim_1b = (const float*)d_in[32];
  const float* tim_1w = (const float*)d_in[33];
  const float* tim_2b = (const float*)d_in[34];
  const float* tim_2w = (const float*)d_in[35];
  const float* wk_bhh = (const float*)d_in[36];
  const float* wk_bih = (const float*)d_in[37];
  const float* wk_ob  = (const float*)d_in[38];
  const float* wk_ow  = (const float*)d_in[39];
  const float* wk_whh = (const float*)d_in[40];
  const float* wk_wih = (const float*)d_in[41];
  const float* wo_bhh = (const float*)d_in[42];
  const float* wo_bih = (const float*)d_in[43];
  const float* wo_ob  = (const float*)d_in[44];
  const float* wo_ow  = (const float*)d_in[45];
  const float* wo_whh = (const float*)d_in[46];
  const float* wo_wih = (const float*)d_in[47];

  float* ws = (float*)d_ws;
  float* temp_pred = ws;                                   // B*2
  float* g_geo  = ws + (size_t)B_SZ*2;                     // B*8 each
  float* g_link = g_geo  + (size_t)B_SZ*NG;
  float* g_time = g_link + (size_t)B_SZ*NG;
  float* n_base = g_time + (size_t)B_SZ*NG;
  const size_t NSZ = (size_t)B_SZ*NNB*NG;                  // 2,293,760 floats
  float* n_geo  = n_base;
  float* n_link = n_geo  + NSZ;
  float* n_time = n_link + NSZ;
  float* h1 = n_base;                                      // reuse after attn
  float* h2 = h1 + (size_t)B_SZ*256;

  // 1. temporal branch
  gru_kernel<<<B_SZ/256, 256, 0, stream>>>(his,
      wk_wih, wk_whh, wk_bih, wk_bhh, wk_ow, wk_ob,
      wo_wih, wo_whh, wo_bih, wo_bhh, wo_ow, wo_ob,
      tf_w, tf_b, temp_pred);

  // 2. neighbor aligns (WMMA): M = B*35 = 286720 rows -> 17920 tiles, 4 waves/block
  const int alignBlocks = (B_SZ*NNB/16) / 4;               // 4480
  align_neigh_kernel<<<alignBlocks, 128, 0, stream>>>(nb_geo,  geo_aw, geo_ab, n_geo);
  align_neigh_kernel<<<alignBlocks, 128, 0, stream>>>(nb_link, lnk_aw, lnk_ab, n_link);
  align_neigh_kernel<<<alignBlocks, 128, 0, stream>>>(nb_time, tim_aw, tim_ab, n_time);

  // 3. attention / aggregation per graph
  graph_attn_kernel<<<B_SZ/256, 256, 0, stream>>>(x_od, geo_aw, geo_ab, geo_1w, geo_1b, geo_2w, geo_2b, n_geo,  g_geo);
  graph_attn_kernel<<<B_SZ/256, 256, 0, stream>>>(x_od, lnk_aw, lnk_ab, lnk_1w, lnk_1b, lnk_2w, lnk_2b, n_link, g_link);
  graph_attn_kernel<<<B_SZ/256, 256, 0, stream>>>(x_od, tim_aw, tim_ab, tim_1w, tim_1b, tim_2w, tim_2b, n_time, g_time);

  // 4. fc1: grid = 128 mt-groups * 4 n-groups; 4 waves/block share the N-group
  fc1_kernel<<<512, 128, 0, stream>>>(x_od, g_time, g_link, g_geo, fc1_w, fc1_b, h1);

  // 5. fc2: N=128 -> 2 groups; 512*2 = 1024 waves -> 256 blocks
  fcrelu_kernel<<<256, 128, 0, stream>>>(h1, fc2_w, fc2_b, h2, 256, 128, 2);

  // 6. fc3 + gating + blend
  head_kernel<<<B_SZ/256, 256, 0, stream>>>(h2, fc3_w, fc3_b, age,
      g1_w, g1_b, g2_w, g2_b, temp_pred, (float*)d_out);
}
